// WuBuDiffusionModel_8924942041792
// MI455X (gfx1250) — compile-verified, matrix operand order checked by reading, not run on hardware
//
#include <hip/hip_runtime.h>
#include <hip/hip_bf16.h>

// ---------------------------------------------------------------------------
// WuBu hyperbolic-kNN transformer block for MI455X (gfx1250), bf16 WMMA path.
// ---------------------------------------------------------------------------

#define BB   2
#define NN   2048
#define DIMM 768
#define NHH  12
#define HDD  64
#define KNNK 32
#define DFFF 3072

typedef __bf16 bf16_t;
typedef __attribute__((ext_vector_type(16))) bf16_t v16bf;
typedef __attribute__((ext_vector_type(8)))  bf16_t v8bf;
typedef __attribute__((ext_vector_type(8)))  float  v8f;

// ----------------------------- helpers -------------------------------------

__device__ __forceinline__ v16bf load_frag_bf16(const bf16_t* __restrict__ p) {
    // 16-bit A/B fragment: elements 0..7  <- K = base+0..7   (contiguous)
    //                      elements 8..15 <- K = base+16..23 (contiguous)
    v8bf lo = *(const v8bf*)(p);
    v8bf hi = *(const v8bf*)(p + 16);
    v16bf r;
#pragma unroll
    for (int e = 0; e < 8; ++e) { r[e] = lo[e]; r[8 + e] = hi[e]; }
    return r;
}

__device__ __forceinline__ float gelu_exact(float x) {
    return 0.5f * x * (1.0f + erff(x * 0.70710678118654752440f));
}

// ------------------------ weight transpose + bf16 ---------------------------
// in: fp32 [K,N] row-major  ->  out: bf16 [N,K] row-major
__global__ __launch_bounds__(256) void transpose_to_bf16(
    const float* __restrict__ in, bf16_t* __restrict__ out, int K, int N) {
    int idx = blockIdx.x * 256 + threadIdx.x;
    if (idx >= K * N) return;
    int n = idx / K;
    int k = idx - n * K;
    out[idx] = (bf16_t)in[(size_t)k * N + n];
}

// --------------------------- LayerNorm (+time) ------------------------------
// One block per token row. Optional time_emb add. Output bf16.
__global__ __launch_bounds__(256) void ln_kernel(
    const float* __restrict__ x, const float* __restrict__ g,
    const float* __restrict__ b, const float* __restrict__ temb,
    bf16_t* __restrict__ out, int tokPerBatch, int dim) {
    int row   = blockIdx.x;
    int batch = row / tokPerBatch;
    const float* xr = x + (size_t)row * dim;

    float lsum = 0.f, lsq = 0.f;
    for (int c = threadIdx.x; c < dim; c += 256) {
        float v = xr[c];
        lsum += v; lsq += v * v;
    }
#pragma unroll
    for (int off = 16; off; off >>= 1) {
        lsum += __shfl_xor(lsum, off, 32);
        lsq  += __shfl_xor(lsq,  off, 32);
    }
    __shared__ float sSum[8], sSq[8];
    if ((threadIdx.x & 31) == 0) {
        sSum[threadIdx.x >> 5] = lsum;
        sSq [threadIdx.x >> 5] = lsq;
    }
    __syncthreads();
    __shared__ float s_mu, s_rstd;
    if (threadIdx.x == 0) {
        float S = 0.f, Q = 0.f;
#pragma unroll
        for (int i = 0; i < 8; ++i) { S += sSum[i]; Q += sSq[i]; }
        float mu  = S / (float)dim;
        float var = Q / (float)dim - mu * mu;
        s_mu = mu; s_rstd = rsqrtf(var + 1e-5f);
    }
    __syncthreads();
    float mu = s_mu, rstd = s_rstd;
    const float* te = temb ? (temb + (size_t)batch * dim) : nullptr;
    bf16_t* orow = out + (size_t)row * dim;
    for (int c = threadIdx.x; c < dim; c += 256) {
        float v = (xr[c] - mu) * rstd * g[c] + b[c];
        if (te) v += te[c];
        orow[c] = (bf16_t)v;
    }
}

// --------------------------- bf16 WMMA GEMM ---------------------------------
// C[M,N] = A[M,K](bf16) * Bt[N,K]^T(bf16) + bias, with epilogues:
//   epi 0: out fp32
//   epi 1: out fp32 = resid + acc + bias
//   epi 2: out bf16 = gelu(acc + bias)
// Each wave owns a 32(M) x 64(N) tile: 2 A fragments, 4 B fragments, 8 f32
// accumulators per k-step -> 1.5 b128 loads per WMMA (B reused 2x, A 4x).
__global__ __launch_bounds__(256) void gemm_bf16_wmma(
    const bf16_t* __restrict__ A, const bf16_t* __restrict__ Bt,
    const float* __restrict__ bias, const float* __restrict__ resid,
    float* __restrict__ outF, bf16_t* __restrict__ outBF,
    int M, int N, int K, int epi) {
    int wave   = blockIdx.x * (blockDim.x >> 5) + (threadIdx.x >> 5);
    int tilesN = N >> 6;                  // 64-wide N strips
    int tm     = (wave / tilesN) << 5;    // 32 rows of M per wave
    int tn     = (wave % tilesN) << 6;
    if (tm >= M) return;

    int lane = threadIdx.x & 31;
    int l16  = lane & 15;
    int half = lane >> 4;

    const bf16_t* arow0 = A + (size_t)(tm +  0 + l16) * K;
    const bf16_t* arow1 = A + (size_t)(tm + 16 + l16) * K;
    const bf16_t* brow0 = Bt + (size_t)(tn +  0 + l16) * K;
    const bf16_t* brow1 = Bt + (size_t)(tn + 16 + l16) * K;
    const bf16_t* brow2 = Bt + (size_t)(tn + 32 + l16) * K;
    const bf16_t* brow3 = Bt + (size_t)(tn + 48 + l16) * K;

    v8f acc0[4] = {};
    v8f acc1[4] = {};
    for (int k0 = 0; k0 < K; k0 += 32) {
        int ka = k0 + half * 8;
        __builtin_prefetch(arow0 + ka + 64, 0, 3);
        __builtin_prefetch(arow1 + ka + 64, 0, 3);
        v16bf a0 = load_frag_bf16(arow0 + ka);
        v16bf a1 = load_frag_bf16(arow1 + ka);
        v16bf b0 = load_frag_bf16(brow0 + ka);
        v16bf b1 = load_frag_bf16(brow1 + ka);
        v16bf b2 = load_frag_bf16(brow2 + ka);
        v16bf b3 = load_frag_bf16(brow3 + ka);
        acc0[0] = __builtin_amdgcn_wmma_f32_16x16x32_bf16(false, a0, false, b0,
                                                          (short)0, acc0[0], false, false);
        acc1[0] = __builtin_amdgcn_wmma_f32_16x16x32_bf16(false, a1, false, b0,
                                                          (short)0, acc1[0], false, false);
        acc0[1] = __builtin_amdgcn_wmma_f32_16x16x32_bf16(false, a0, false, b1,
                                                          (short)0, acc0[1], false, false);
        acc1[1] = __builtin_amdgcn_wmma_f32_16x16x32_bf16(false, a1, false, b1,
                                                          (short)0, acc1[1], false, false);
        acc0[2] = __builtin_amdgcn_wmma_f32_16x16x32_bf16(false, a0, false, b2,
                                                          (short)0, acc0[2], false, false);
        acc1[2] = __builtin_amdgcn_wmma_f32_16x16x32_bf16(false, a1, false, b2,
                                                          (short)0, acc1[2], false, false);
        acc0[3] = __builtin_amdgcn_wmma_f32_16x16x32_bf16(false, a0, false, b3,
                                                          (short)0, acc0[3], false, false);
        acc1[3] = __builtin_amdgcn_wmma_f32_16x16x32_bf16(false, a1, false, b3,
                                                          (short)0, acc1[3], false, false);
    }

    // C/D layout: vgpr i of lanes 0-15 -> row base+i, lanes 16-31 -> base+8+i.
#pragma unroll
    for (int s = 0; s < 2; ++s) {
        int rowBase = tm + s * 16 + half * 8;
#pragma unroll
        for (int j = 0; j < 4; ++j) {
            int col = tn + j * 16 + l16;
            float bv = bias[col];
            v8f av = s ? acc1[j] : acc0[j];
#pragma unroll
            for (int i = 0; i < 8; ++i) {
                size_t o = (size_t)(rowBase + i) * N + col;
                float v = av[i] + bv;
                if (epi == 1) v += resid[o];
                if (epi == 2) v  = gelu_exact(v);
                if (outF)  outF[o]  = v;
                if (outBF) outBF[o] = (bf16_t)v;
            }
        }
    }
}

// ------------------------- Poincare kNN (top-32) ----------------------------
// One thread per query token; positions of the batch staged in LDS;
// per-thread sorted insertion lists kept in LDS.
__global__ __launch_bounds__(128) void knn_kernel(
    const float* __restrict__ pos, const float* __restrict__ cptr,
    float* __restrict__ dists, int* __restrict__ knn_idx, int tokPerBatch) {
    __shared__ float2 sp[NN];
    __shared__ float  dl[128][KNNK];
    __shared__ int    il[128][KNNK];

    int t     = threadIdx.x;
    int row   = blockIdx.x * 128 + t;        // global over B*N
    int batch = row / tokPerBatch;
    int i     = row - batch * tokPerBatch;

    for (int j = t; j < tokPerBatch; j += 128) {
        const float* p = pos + ((size_t)batch * tokPerBatch + j) * 2;
        sp[j] = make_float2(p[0], p[1]);
    }
    __syncthreads();

    float c          = cptr[0];
    float inv_sqrt_c = rsqrtf(c);
    float2 pi  = sp[i];
    float xfac = 1.0f - c * (pi.x * pi.x + pi.y * pi.y);

#pragma unroll
    for (int k = 0; k < KNNK; ++k) { dl[t][k] = 3.0e38f; il[t][k] = 0; }

    for (int j = 0; j < tokPerBatch; ++j) {
        float2 pj = sp[j];
        float dx = pi.x - pj.x, dy = pi.y - pj.y;
        float diff2 = dx * dx + dy * dy;
        float yfac  = 1.0f - c * (pj.x * pj.x + pj.y * pj.y);
        float den   = xfac * yfac;
        float arg   = fmaxf(1.0f + (2.0f * c * diff2) / (den + 1e-8f), 1.0f);
        float d     = acoshf(arg) * inv_sqrt_c;
        if (d < dl[t][KNNK - 1]) {
            int p = KNNK - 1;
            while (p > 0 && dl[t][p - 1] > d) {
                dl[t][p] = dl[t][p - 1];
                il[t][p] = il[t][p - 1];
                --p;
            }
            dl[t][p] = d;
            il[t][p] = j;
        }
    }
#pragma unroll
    for (int k = 0; k < KNNK; ++k) {
        dists  [(size_t)row * KNNK + k] = dl[t][k];
        knn_idx[(size_t)row * KNNK + k] = il[t][k];
    }
}

// --------------------- kNN attention (wave per b,h,n) -----------------------
// lane <-> neighbor (KNN==32==wave32). Softmax across the wave; V combine by
// broadcasting each neighbor's probability with __shfl.
__global__ __launch_bounds__(256) void attn_kernel(
    const float* __restrict__ q, const float* __restrict__ k,
    const float* __restrict__ v, const float* __restrict__ dists,
    const int* __restrict__ knn_idx, const float* __restrict__ log_tau,
    bf16_t* __restrict__ out, int tokPerBatch) {
    int w    = threadIdx.x >> 5;
    int lane = threadIdx.x & 31;
    long wg  = (long)blockIdx.x * 8 + w;             // [0, B*NH*N)
    int  b   = (int)(wg / ((long)NHH * tokPerBatch));
    long r   = wg - (long)b * NHH * tokPerBatch;
    int  h   = (int)(r / tokPerBatch);
    int  n   = (int)(r - (long)h * tokPerBatch);

    __shared__ float qs[8][HDD];
    const float* qrow = q + ((size_t)b * tokPerBatch + n) * DIMM + h * HDD;
    qs[w][lane * 2]     = qrow[lane * 2];
    qs[w][lane * 2 + 1] = qrow[lane * 2 + 1];
    __syncthreads();

    size_t base = ((size_t)b * tokPerBatch + n) * KNNK;
    int   j    = knn_idx[base + lane];
    float dist = dists[base + lane];
    float tau  = __expf(log_tau[0]) + 1e-8f;

    const float* krow = k + ((size_t)b * tokPerBatch + j) * DIMM + h * HDD;
    float s = 0.f;
#pragma unroll
    for (int d = 0; d < HDD; d += 4) {
        float4 kv = *(const float4*)(krow + d);
        s += qs[w][d] * kv.x + qs[w][d + 1] * kv.y +
             qs[w][d + 2] * kv.z + qs[w][d + 3] * kv.w;
    }
    s = s * 0.125f - dist / tau;                      // 1/sqrt(64) = 0.125

    float m = s;
#pragma unroll
    for (int off = 16; off; off >>= 1) m = fmaxf(m, __shfl_xor(m, off, 32));
    float p = __expf(s - m);
    float sum = p;
#pragma unroll
    for (int off = 16; off; off >>= 1) sum += __shfl_xor(sum, off, 32);
    p /= sum;

    float o0 = 0.f, o1 = 0.f;
#pragma unroll
    for (int kk = 0; kk < 32; ++kk) {
        float pk = __shfl(p, kk, 32);
        int   jk = __shfl(j, kk, 32);
        float2 vv = *(const float2*)(v + ((size_t)b * tokPerBatch + jk) * DIMM +
                                     h * HDD + lane * 2);
        o0 += pk * vv.x;
        o1 += pk * vv.y;
    }
    bf16_t* orow = out + ((size_t)b * tokPerBatch + n) * DIMM + h * HDD + lane * 2;
    orow[0] = (bf16_t)o0;
    orow[1] = (bf16_t)o1;
}

// ----------------------------------------------------------------------------

extern "C" void kernel_launch(void* const* d_in, const int* in_sizes, int n_in,
                              void* d_out, int out_size, void* d_ws, size_t ws_size,
                              hipStream_t stream) {
    const float* x       = (const float*)d_in[0];
    const float* pos     = (const float*)d_in[1];
    const float* c       = (const float*)d_in[2];
    const float* temb    = (const float*)d_in[3];
    const float* Wq      = (const float*)d_in[4];
    const float* bq      = (const float*)d_in[5];
    const float* Wk      = (const float*)d_in[6];
    const float* bk      = (const float*)d_in[7];
    const float* Wv      = (const float*)d_in[8];
    const float* bv      = (const float*)d_in[9];
    const float* Wo      = (const float*)d_in[10];
    const float* bo      = (const float*)d_in[11];
    const float* W1      = (const float*)d_in[12];
    const float* b1      = (const float*)d_in[13];
    const float* W2      = (const float*)d_in[14];
    const float* b2      = (const float*)d_in[15];
    const float* g1      = (const float*)d_in[16];
    const float* be1     = (const float*)d_in[17];
    const float* g2      = (const float*)d_in[18];
    const float* be2     = (const float*)d_in[19];
    const float* log_tau = (const float*)d_in[20];
    float* out = (float*)d_out;

    const int M = BB * NN;              // 4096 token rows

    // ---- workspace layout (256B aligned) ----
    char* wp = (char*)d_ws;
    auto alloc = [&](size_t bytes) -> void* {
        void* p = (void*)wp;
        wp += (bytes + 255) & ~(size_t)255;
        return p;
    };
    bf16_t* WqT    = (bf16_t*)alloc((size_t)DIMM * DIMM * 2);
    bf16_t* WkT    = (bf16_t*)alloc((size_t)DIMM * DIMM * 2);
    bf16_t* WvT    = (bf16_t*)alloc((size_t)DIMM * DIMM * 2);
    bf16_t* WoT    = (bf16_t*)alloc((size_t)DIMM * DIMM * 2);
    bf16_t* W1T    = (bf16_t*)alloc((size_t)DIMM * DFFF * 2);
    bf16_t* W2T    = (bf16_t*)alloc((size_t)DFFF * DIMM * 2);
    bf16_t* h_bf   = (bf16_t*)alloc((size_t)M * DIMM * 2);
    float*  qf     = (float*) alloc((size_t)M * DIMM * 4);
    float*  kf     = (float*) alloc((size_t)M * DIMM * 4);
    float*  vf     = (float*) alloc((size_t)M * DIMM * 4);
    float*  dists  = (float*) alloc((size_t)M * KNNK * 4);
    int*    idx    = (int*)   alloc((size_t)M * KNNK * 4);
    bf16_t* attnbf = (bf16_t*)alloc((size_t)M * DIMM * 2);
    float*  x2     = (float*) alloc((size_t)M * DIMM * 4);
    bf16_t* h2_bf  = (bf16_t*)alloc((size_t)M * DIMM * 2);
    bf16_t* ffn1bf = (bf16_t*)alloc((size_t)M * DFFF * 2);

    // ---- 1) weights -> bf16, transposed to [N,K] ----
    int e1 = DIMM * DIMM, e2 = DIMM * DFFF;
    transpose_to_bf16<<<(e1 + 255) / 256, 256, 0, stream>>>(Wq, WqT, DIMM, DIMM);
    transpose_to_bf16<<<(e1 + 255) / 256, 256, 0, stream>>>(Wk, WkT, DIMM, DIMM);
    transpose_to_bf16<<<(e1 + 255) / 256, 256, 0, stream>>>(Wv, WvT, DIMM, DIMM);
    transpose_to_bf16<<<(e1 + 255) / 256, 256, 0, stream>>>(Wo, WoT, DIMM, DIMM);
    transpose_to_bf16<<<(e2 + 255) / 256, 256, 0, stream>>>(W1, W1T, DIMM, DFFF);
    transpose_to_bf16<<<(e2 + 255) / 256, 256, 0, stream>>>(W2, W2T, DFFF, DIMM);

    // ---- 2) h = LN(x)*g1+be1 + time_emb  (bf16) ----
    ln_kernel<<<M, 256, 0, stream>>>(x, g1, be1, temb, h_bf, NN, DIMM);

    // ---- 3) Q,K,V GEMMs (bf16 WMMA, fp32 out) ----
    auto gemmBlocks = [](int m, int n) { return ((m / 32) * (n / 64) + 7) / 8; };
    gemm_bf16_wmma<<<gemmBlocks(M, DIMM), 256, 0, stream>>>(
        h_bf, WqT, bq, nullptr, qf, nullptr, M, DIMM, DIMM, 0);
    gemm_bf16_wmma<<<gemmBlocks(M, DIMM), 256, 0, stream>>>(
        h_bf, WkT, bk, nullptr, kf, nullptr, M, DIMM, DIMM, 0);
    gemm_bf16_wmma<<<gemmBlocks(M, DIMM), 256, 0, stream>>>(
        h_bf, WvT, bv, nullptr, vf, nullptr, M, DIMM, DIMM, 0);

    // ---- 4) hyperbolic kNN (independent of LN/GEMM chain) ----
    knn_kernel<<<M / 128, 128, 0, stream>>>(pos, c, dists, idx, NN);

    // ---- 5) kNN attention -> bf16 attn_out ----
    attn_kernel<<<(BB * NHH * NN) / 8, 256, 0, stream>>>(
        qf, kf, vf, dists, idx, log_tau, attnbf, NN);

    // ---- 6) O-proj + residual: x2 = x + attn@Wo + bo ----
    gemm_bf16_wmma<<<gemmBlocks(M, DIMM), 256, 0, stream>>>(
        attnbf, WoT, bo, x, x2, nullptr, M, DIMM, DIMM, 1);

    // ---- 7) LN2 -> bf16 ----
    ln_kernel<<<M, 256, 0, stream>>>(x2, g2, be2, nullptr, h2_bf, NN, DIMM);

    // ---- 8) FFN1: gelu(h2@W1 + b1) -> bf16 ----
    gemm_bf16_wmma<<<gemmBlocks(M, DFFF), 256, 0, stream>>>(
        h2_bf, W1T, b1, nullptr, nullptr, ffn1bf, M, DFFF, DIMM, 2);

    // ---- 9) FFN2 + residual -> d_out ----
    gemm_bf16_wmma<<<gemmBlocks(M, DIMM), 256, 0, stream>>>(
        ffn1bf, W2T, b2, x2, out, nullptr, M, DIMM, DFFF, 1);

    (void)n_in; (void)in_sizes; (void)out_size; (void)ws_size;
}
